// SparseConvUNet_18906446037375
// MI455X (gfx1250) — matrix operand order repaired and checked by reading, not compiled, not run on hardware
//
#include <hip/hip_runtime.h>
#include <stdint.h>

// ---------------------------------------------------------------------------
// SparseConvUNet forward for gfx1250 (MI455X, wave32, WMMA).
// Core motif: gather-GEMM-accumulate via v_wmma_f32_16x16x32_f16 with
// double-buffered async (ASYNCcnt) LDS staging of the weight panels.
// ---------------------------------------------------------------------------

#define NFINE 65536
#define EPSBN 1e-4f

typedef __attribute__((ext_vector_type(16))) _Float16 v16h;
typedef __attribute__((ext_vector_type(8)))  _Float16 v8h;
typedef __attribute__((ext_vector_type(8)))  float    v8f;

// ---------------------------------------------------------------- utilities
__global__ void k_write_int(int* p, int v) {
  if (threadIdx.x == 0 && blockIdx.x == 0) p[0] = v;
}

__global__ void k_fill_f32(float* p, size_t n) {
  size_t t = (size_t)blockIdx.x * blockDim.x + threadIdx.x;
  if (t < n) p[t] = 0.f;
}

// generic strided column copy (also used as flat copy when strides match)
__global__ void k_copy_cols(const float* __restrict__ s, int ss,
                            float* __restrict__ d, int ds,
                            int nc, size_t rows) {
  size_t t = (size_t)blockIdx.x * blockDim.x + threadIdx.x;
  if (t >= rows * (size_t)nc) return;
  size_t r = t / (size_t)nc;
  int    c = (int)(t % (size_t)nc);
  d[r * (size_t)ds + c] = s[r * (size_t)ss + c];
}

__global__ void k_add_bias(float* __restrict__ o, const float* __restrict__ b,
                           int C, size_t total) {
  size_t t = (size_t)blockIdx.x * blockDim.x + threadIdx.x;
  if (t < total) o[t] += b[t % (size_t)C];
}

// ---------------------------------------------------------------- batchnorm
// partial sums: blockDim.x == C (multiple of 32, <=256); 256 rows per block
__global__ void k_bn_partial(const float* __restrict__ X, int C,
                             const int* __restrict__ nd,
                             float* __restrict__ st) {
  int c = threadIdx.x;
  int n = nd[0];
  size_t r0 = (size_t)blockIdx.x * 256;
  float s = 0.f, q = 0.f;
  for (int j = 0; j < 256; ++j) {
    size_t r = r0 + (size_t)j;
    if (r < (size_t)n) {
      float v = X[r * (size_t)C + c];
      s += v; q += v * v;
    }
  }
  atomicAdd(&st[c], s);
  atomicAdd(&st[C + c], q);
}

__global__ void k_bn_finalize(const float* __restrict__ st,
                              const float* __restrict__ g,
                              const float* __restrict__ b,
                              const int* __restrict__ nd, int C,
                              float* __restrict__ sb) {
  int c = threadIdx.x;
  if (c >= C) return;
  float n = (float)nd[0];
  float m = st[c] / n;
  float v = st[C + c] / n - m * m;
  float s = g[c] * rsqrtf(v + EPSBN);
  sb[c]     = s;
  sb[C + c] = b[c] - m * s;
}

// apply (optional) BN scale/shift + ReLU, convert to f16, zero-pad K dim and
// all rows >= n (row n is the zero pad row used for missing neighbors).
__global__ void k_cvt_f16(const float* __restrict__ X, int C,
                          const float* __restrict__ sb,
                          const int* __restrict__ nd,
                          _Float16* __restrict__ H, int Kpad, size_t rows) {
  size_t t = (size_t)blockIdx.x * blockDim.x + threadIdx.x;
  if (t >= rows * (size_t)Kpad) return;
  size_t r = t / (size_t)Kpad;
  int    c = (int)(t % (size_t)Kpad);
  int n = nd[0];
  float v = 0.f;
  if (r < (size_t)n && c < C) {
    v = X[r * (size_t)C + c];
    if (sb) v = fmaxf(v * sb[c] + sb[C + c], 0.f);
  }
  H[r * (size_t)Kpad + c] = (_Float16)v;
}

// ------------------------------------------------- weight repack to B-frags
// dst layout: [tap][kc][ntile][lane][16] f16 so that each lane's B fragment
// for v_wmma_f32_16x16x32_f16 is 32 contiguous bytes.
// lane<16 : col=lane,    K = kc*32 + 0..15
// lane>=16: col=lane-16, K = kc*32 + 16..31
__global__ void k_repack_wfrag(const float* __restrict__ W, int taps, int Cin,
                               int Kpad, int Cout, _Float16* __restrict__ dst) {
  size_t t0 = (size_t)blockIdx.x * blockDim.x + threadIdx.x;
  size_t total = (size_t)taps * (size_t)Kpad * (size_t)Cout;
  if (t0 >= total) return;
  int    j   = (int)(t0 & 15);
  size_t r   = t0 >> 4;
  int  lane  = (int)(r & 31); r >>= 5;
  int  ntls  = Cout >> 4;
  int  nt    = (int)(r % (size_t)ntls); r /= (size_t)ntls;
  int  nkc   = Kpad >> 5;
  int  kc    = (int)(r % (size_t)nkc);
  int  tap   = (int)(r / (size_t)nkc);
  int  K     = kc * 32 + (lane >> 4) * 16 + j;
  int  col   = nt * 16 + (lane & 15);
  float v = (K < Cin) ? W[((size_t)tap * Cin + K) * (size_t)Cout + col] : 0.f;
  dst[t0] = (_Float16)v;
}

// -------------------------------------------------- core: gather-GEMM WMMA
// 128 threads = 4 wave32; each wave computes one 16x16 (rows x outcols) tile,
// accumulating over `taps` rulebook offsets and Kpad/32 K-chunks.
// The per-tap B panel (shared by all 4 waves) is staged into LDS with
// global_load_async_to_lds_b128, double-buffered so tap t+1's DMA overlaps
// tap t's WMMA compute (pay only s_wait_asynccnt + barrier per tap).
// nb == nullptr -> identity gather (1x1 conv / linear).
// Ci != nullptr -> fused residual init of the accumulator.
__global__ __launch_bounds__(128)
void k_wmma_gather(const _Float16* __restrict__ Xh, int Kpad,
                   const _Float16* __restrict__ Wf,
                   const int* __restrict__ nb, int taps,
                   const float* __restrict__ Ci,
                   float* __restrict__ Out, int Cout,
                   const int* __restrict__ nd) {
  __shared__ int sidx[2][64];
  __shared__ __align__(16) _Float16 sB[2][8 * 512];  // 2 x (<=8 K-chunk) panels
  const int n = nd[0];
  if ((int)blockIdx.x * 64 >= n) return;  // uniform early-out per block

  const int lane    = threadIdx.x & 31;
  const int wave    = threadIdx.x >> 5;
  const int mrow    = ((int)blockIdx.x * 4 + wave) * 16;
  const int colbase = (int)blockIdx.y * 16;
  const int nkc     = Kpad >> 5;
  const int ntiles  = Cout >> 4;
  const int mh      = lane >> 4;   // half-wave select
  const int m       = lane & 15;

  // stage tap t's neighbor indices + B-fragment panel into LDS buffer t&1.
  // B staging is asynchronous (tracked by ASYNCcnt, waited at the consumer).
  auto stage = [&](int t) {
    if (threadIdx.x < 64) {
      int row = (int)blockIdx.x * 64 + (int)threadIdx.x;
      int idx;
      if (row < n) idx = nb ? nb[(size_t)row * taps + t] : row;
      else         idx = n;                  // zero pad row
      sidx[t & 1][threadIdx.x] = idx;
      // warm L2/WGP$ for the scattered A-row gathers of this tap
      __builtin_prefetch(Xh + (size_t)idx * Kpad, 0, 0);
    }
    const int chunks = nkc << 6;             // 16-byte chunks in the panel
    const uint32_t lbase = (uint32_t)(uintptr_t)&sB[t & 1][0];
    for (int c = threadIdx.x; c < chunks; c += 128) {
      int kc = c >> 6, wi = c & 63;
      const _Float16* src = Wf +
          (((size_t)t * nkc + kc) * ntiles + (size_t)blockIdx.y) * 512 +
          (size_t)wi * 8;
      uint32_t ldsoff = lbase + (uint32_t)c * 16u;
      asm volatile("global_load_async_to_lds_b128 %0, %1, off"
                   :: "v"(ldsoff), "v"((uint64_t)(uintptr_t)src)
                   : "memory");
    }
  };

  v8f acc = {};
  if (Ci) {
#pragma unroll
    for (int r = 0; r < 8; ++r) {
      int row = mrow + r + mh * 8;           // C/D layout: VGPR r -> M=r / r+8
      acc[r] = (row < n) ? Ci[(size_t)row * Cout + colbase + m] : 0.f;
    }
  }

  // pipeline prologue
  stage(0);
  asm volatile("s_wait_asynccnt 0x0" ::: "memory");
  __syncthreads();

  for (int t = 0; t < taps; ++t) {
    if (t + 1 < taps) stage(t + 1);          // async: overlaps compute below

    const int gidx = sidx[t & 1][wave * 16 + m];
    // A k order used here (contiguous, same as B repack): lanes 0-15 hold K {0..15},
    // lanes 16-31 hold K {16..31}  -> two 16B chunks at +0 and +8.
    const _Float16* arow = Xh + (size_t)gidx * Kpad + mh * 16;
    const _Float16* bpan = &sB[t & 1][0];
    for (int kc = 0; kc < nkc; ++kc) {
      v8h c0 = *(const v8h*)(arow + kc * 32);
      v8h c1 = *(const v8h*)(arow + kc * 32 + 8);
      v16h a;
#pragma unroll
      for (int j = 0; j < 8; ++j) { a[j] = c0[j]; a[8 + j] = c1[j]; }
      v16h b = *(const v16h*)(bpan + ((size_t)kc * 32 + lane) * 16);
      acc = __builtin_amdgcn_wmma_f32_16x16x32_f16(
          false, a, false, b, (short)0, acc, false, false);
    }

    if (t + 1 < taps) {
      asm volatile("s_wait_asynccnt 0x0" ::: "memory");
      __syncthreads();
    }
  }

#pragma unroll
  for (int r = 0; r < 8; ++r) {
    int row = mrow + r + mh * 8;
    if (row < n) Out[(size_t)row * Cout + colbase + m] = acc[r];
  }
}

// ---------------------------------------------- stride-2 down / up convs
// per-row tap selection breaks WMMA tiling; these are <1% of total FLOPs.
__global__ void k_downconv(const _Float16* __restrict__ Y, int Cin,
                           const float* __restrict__ W,
                           const int* __restrict__ inv,
                           const int* __restrict__ off,
                           const int* __restrict__ nd,
                           float* __restrict__ Out, int Cout) {
  size_t t = (size_t)blockIdx.x * blockDim.x + threadIdx.x;
  size_t r = t / (size_t)Cout;
  int    c = (int)(t % (size_t)Cout);
  int n = nd[0];
  if (r >= (size_t)n) return;
  const _Float16* y = Y + r * (size_t)Cin;      // Kpad == Cin (mult of 32)
  const float*    w = W + (size_t)off[r] * Cin * Cout + c;
  float acc = 0.f;
  for (int k = 0; k < Cin; ++k) acc += (float)y[k] * w[(size_t)k * Cout];
  atomicAdd(&Out[(size_t)inv[r] * Cout + c], acc);
}

__global__ void k_deconv(const _Float16* __restrict__ Yc, int Cin,
                         const float* __restrict__ W,
                         const int* __restrict__ inv,
                         const int* __restrict__ off,
                         const int* __restrict__ nd,
                         float* __restrict__ Out, int ostride, int col0,
                         int Cout) {
  size_t t = (size_t)blockIdx.x * blockDim.x + threadIdx.x;
  size_t r = t / (size_t)Cout;
  int    c = (int)(t % (size_t)Cout);
  int n = nd[0];
  if (r >= (size_t)n) return;
  const _Float16* y = Yc + (size_t)inv[r] * Cin;
  const float*    w = W + (size_t)off[r] * Cin * Cout + c;
  float acc = 0.f;
  for (int k = 0; k < Cin; ++k) acc += (float)y[k] * w[(size_t)k * Cout];
  Out[r * (size_t)ostride + col0 + c] = acc;
}

// ---------------------------------------------------------------------------
extern "C" void kernel_launch(void* const* d_in, const int* in_sizes, int n_in,
                              void* d_out, int out_size, void* d_ws,
                              size_t ws_size, hipStream_t stream) {
  (void)in_sizes; (void)n_in; (void)out_size; (void)ws_size;
  static const int PLANES[5] = {32, 64, 96, 128, 160};

  const float* feats = (const float*)d_in[0];
  const float* P     = (const float*)d_in[1];
  const int* nb[5]   = {(const int*)d_in[2], (const int*)d_in[3],
                        (const int*)d_in[4], (const int*)d_in[5],
                        (const int*)d_in[6]};
  const int* dmap[4] = {(const int*)d_in[7],  (const int*)d_in[10],
                        (const int*)d_in[13], (const int*)d_in[16]};
  const int* omap[4] = {(const int*)d_in[8],  (const int*)d_in[11],
                        (const int*)d_in[14], (const int*)d_in[17]};
  const int* ncnt[4] = {(const int*)d_in[9],  (const int*)d_in[12],
                        (const int*)d_in[15], (const int*)d_in[18]};

  // ---- param offsets (JAX pytree flatten = sorted dict keys) ----
  size_t po = 0;
  auto take = [&](size_t c) { size_t o = po; po += c; return o; };
  struct Blk { size_t b1, b2, g1, g2, nin, w1, w2; };
  auto mkblk = [&](int cin, int cout, bool nin) {
    Blk B;  // key order: b1 < b2 < g1 < g2 < nin < w1 < w2
    B.b1 = take((size_t)cin);  B.b2 = take((size_t)cout);
    B.g1 = take((size_t)cin);  B.g2 = take((size_t)cout);
    B.nin = nin ? take((size_t)cin * cout) : 0;
    B.w1 = take(27ull * cin * cout);
    B.w2 = take(27ull * cout * cout);
    return B;
  };
  struct DU { size_t b, g, w; };
  // top-level sorted: deep, down0..3, final_bn, lin_b, lin_w, post0..3,
  //                   pre0..3, stem, up0..3
  Blk Bdeep = mkblk(160, 160, false);
  DU down[4];
  for (int i = 0; i < 4; ++i) {
    down[i].b = take((size_t)PLANES[i]);
    down[i].g = take((size_t)PLANES[i]);
    down[i].w = take(8ull * PLANES[i] * PLANES[i + 1]);
  }
  size_t fb_b = take(32), fb_g = take(32);
  size_t lin_b = take(16), lin_w = take(32 * 16);
  Blk post[4];
  for (int i = 0; i < 4; ++i) post[i] = mkblk(2 * PLANES[i], PLANES[i], true);
  Blk pre[4];
  for (int i = 0; i < 4; ++i) pre[i] = mkblk(PLANES[i], PLANES[i], false);
  size_t stem = take(27ull * 3 * 32);
  DU up[4];
  for (int i = 0; i < 4; ++i) {
    up[i].b = take((size_t)PLANES[i + 1]);
    up[i].g = take((size_t)PLANES[i + 1]);
    up[i].w = take(8ull * PLANES[i + 1] * PLANES[i]);
  }

  // ---- workspace layout ----
  uintptr_t wb = (uintptr_t)d_ws;
  size_t wo = 0;
  auto walloc = [&](size_t bytes) {
    void* p = (void*)(wb + wo);
    wo = (wo + bytes + 255) & ~(size_t)255;
    return p;
  };
  float*    STAT = (float*)walloc(2 * 256 * sizeof(float));
  float*    SB   = (float*)walloc(2 * 256 * sizeof(float));
  int*      NC   = (int*)walloc(64);
  _Float16* WF   = (_Float16*)walloc((size_t)8 << 20);
  _Float16* H16  = (_Float16*)walloc((size_t)(NFINE + 1) * 256 * 2);
  float*    FA   = (float*)walloc((size_t)NFINE * 256 * 4);
  float*    FB   = (float*)walloc((size_t)NFINE * 256 * 4);
  float*    FC   = (float*)walloc((size_t)NFINE * 256 * 4);
  float*    FD   = (float*)walloc((size_t)NFINE * 128 * 4);
  float* SK[4];
  for (int i = 0; i < 4; ++i)
    SK[i] = (float*)walloc((size_t)NFINE * PLANES[i] * 4);

  auto blocks = [](size_t n) { return dim3((unsigned)((n + 255) / 256)); };

  // ---- launch helpers ----
  auto fillz = [&](float* p, size_t c) {
    k_fill_f32<<<blocks(c), 256, 0, stream>>>(p, c);
  };
  auto cvt = [&](const float* x, int C, int Kpad, const float* sb,
                 const int* nd) {
    size_t tot = (size_t)(NFINE + 1) * Kpad;
    k_cvt_f16<<<blocks(tot), 256, 0, stream>>>(x, C, sb, nd, H16, Kpad,
                                               (size_t)(NFINE + 1));
  };
  auto bnsb = [&](const float* x, int C, size_t goff, size_t boff,
                  const int* nd) {
    fillz(STAT, (size_t)2 * C);
    k_bn_partial<<<dim3((NFINE + 255) / 256), dim3((unsigned)C), 0, stream>>>(
        x, C, nd, STAT);
    k_bn_finalize<<<1, dim3((unsigned)C), 0, stream>>>(STAT, P + goff,
                                                       P + boff, nd, C, SB);
  };
  auto repack = [&](size_t woff, int taps, int Cin, int Cout) {
    int Kpad = (Cin + 31) & ~31;
    size_t tot = (size_t)taps * Kpad * Cout;
    k_repack_wfrag<<<blocks(tot), 256, 0, stream>>>(P + woff, taps, Cin, Kpad,
                                                    Cout, WF);
    return Kpad;
  };
  auto wmma = [&](int Kpad, int Cout, const int* nbp, int taps,
                  const float* cinit, float* out, const int* nd) {
    k_wmma_gather<<<dim3((NFINE + 63) / 64, (unsigned)(Cout / 16)), dim3(128),
                    0, stream>>>(H16, Kpad, WF, nbp, taps, cinit, out, Cout,
                                 nd);
  };
  auto resblock = [&](const float* in, int Cin, int Cout, float* out,
                      float* scb, float* yb, const Blk& W, const int* nbp,
                      const int* nd) {
    const float* cinit;
    if (Cin != Cout) {  // NiN shortcut: sc = in @ nin (identity gather)
      cvt(in, Cin, (Cin + 31) & ~31, nullptr, nd);
      int kp = repack(W.nin, 1, Cin, Cout);
      wmma(kp, Cout, nullptr, 1, nullptr, scb, nd);
      cinit = scb;
    } else {
      cinit = in;
    }
    bnsb(in, Cin, W.g1, W.b1, nd);
    cvt(in, Cin, (Cin + 31) & ~31, SB, nd);
    { int kp = repack(W.w1, 27, Cin, Cout);
      wmma(kp, Cout, nbp, 27, nullptr, yb, nd); }
    bnsb(yb, Cout, W.g2, W.b2, nd);
    cvt(yb, Cout, Cout, SB, nd);
    { int kp = repack(W.w2, 27, Cout, Cout);
      wmma(kp, Cout, nbp, 27, cinit, out, nd); }  // fused residual add
  };

  // ---- forward pass ----
  k_write_int<<<1, 1, 0, stream>>>(NC, NFINE);
  const int* ND[5] = {NC, ncnt[0], ncnt[1], ncnt[2], ncnt[3]};

  // stem: 3 -> 32 submanifold conv
  cvt(feats, 3, 32, nullptr, ND[0]);
  { int kp = repack(stem, 27, 3, 32);
    wmma(kp, 32, nb[0], 27, nullptr, FA, ND[0]); }

  // encoder
  for (int i = 0; i < 4; ++i) {
    int a = PLANES[i], b = PLANES[i + 1];
    resblock(FA, a, a, FA, FC, FB, pre[i], nb[i], ND[i]);
    k_copy_cols<<<blocks((size_t)NFINE * a), 256, 0, stream>>>(
        FA, a, SK[i], a, a, (size_t)NFINE);               // save skip
    bnsb(FA, a, down[i].g, down[i].b, ND[i]);
    cvt(FA, a, a, SB, ND[i]);
    fillz(FB, (size_t)NFINE * b);
    k_downconv<<<blocks((size_t)NFINE * b), 256, 0, stream>>>(
        H16, a, P + down[i].w, dmap[i], omap[i], ND[i], FB, b);
    k_copy_cols<<<blocks((size_t)NFINE * b), 256, 0, stream>>>(
        FB, b, FA, b, b, (size_t)NFINE);                  // x = coarse
  }

  // deepest resblock
  resblock(FA, 160, 160, FA, FC, FB, Bdeep, nb[4], ND[4]);

  // decoder
  for (int i = 3; i >= 0; --i) {
    int a = PLANES[i], b = PLANES[i + 1];
    bnsb(FA, b, up[i].g, up[i].b, ND[i + 1]);
    cvt(FA, b, b, SB, ND[i + 1]);
    // concat buffer FC: [skip | deconv(up)]
    k_deconv<<<blocks((size_t)NFINE * a), 256, 0, stream>>>(
        H16, b, P + up[i].w, dmap[i], omap[i], ND[i], FC, 2 * a, a, a);
    k_copy_cols<<<blocks((size_t)NFINE * a), 256, 0, stream>>>(
        SK[i], a, FC, 2 * a, a, (size_t)NFINE);
    resblock(FC, 2 * a, a, FA, FD, FB, post[i], nb[i], ND[i]);
  }

  // final BN-ReLU + linear 32 -> 16 (identity-gather WMMA) + bias
  bnsb(FA, 32, fb_g, fb_b, ND[0]);
  cvt(FA, 32, 32, SB, ND[0]);
  { int kp = repack(lin_w, 1, 32, 16);
    wmma(kp, 16, nullptr, 1, nullptr, (float*)d_out, ND[0]); }
  k_add_bias<<<blocks((size_t)NFINE * 16), 256, 0, stream>>>(
      (float*)d_out, P + lin_b, 16, (size_t)NFINE * 16);
}
